// AF2GNN1_40759239639228
// MI455X (gfx1250) — matrix operand order repaired and checked by reading, not compiled, not run on hardware
//
#include <hip/hip_runtime.h>
#include <hip/hip_bf16.h>
#include <cstddef>

// ---------------------------------------------------------------------------
// AF2GNN on MI455X (gfx1250, wave32). Heavy GEMMs use v_wmma_f32_16x16x32_bf16
// with f32->bf16 conversion in-register for Q (read once per GEMM, 512MB each),
// B operands pre-packed into exact WMMA fragment layout. Graph ops use a
// deterministic CSR neighbor list (adj is 2% sparse, ~41 nnz/row) so the
// adj-matmuls and masked softmaxes loop over ~41 neighbors, not 2048 columns.
// ---------------------------------------------------------------------------

#define BNC 0.9999950000374997f   // 1/sqrt(1+1e-5)
#define CSR_CAP 640               // max neighbors kept per row (mean ~41)

typedef __attribute__((ext_vector_type(16))) __bf16 v16bf;
typedef __attribute__((ext_vector_type(8)))  float  v8f;

__device__ __forceinline__ v8f wmma_bf16(v16bf a, v16bf b, v8f c) {
  // D = A(16x32) * B(32x16) + C, f32 accumulate
  return __builtin_amdgcn_wmma_f32_16x16x32_bf16(
      false, a, false, b, (short)0, c, false, false);
}

__device__ __forceinline__ float lrelu_(float x, float al) {
  return x >= 0.f ? x : al * x;
}

// ---------------- init: zero sums + C1 (contiguous at ws start) -------------
__global__ void k_init(float* w) {
  int i = blockIdx.x * 256 + threadIdx.x;   // grid sized exactly
  w[i] = 0.f;
}

// ---------------- column sums of Q [65536,2048], deterministic --------------
__global__ void k_colsum_p(const float* __restrict__ Q, float* __restrict__ part) {
  int col = blockIdx.x * 256 + threadIdx.x;       // 8 blocks * 256 = 2048 cols
  int r0  = blockIdx.y * 1024;                    // 64 row-chunks
  float s = 0.f;
  for (int r = r0; r < r0 + 1024; ++r)
    s += Q[(size_t)r * 2048 + col];
  part[(size_t)blockIdx.y * 2048 + col] = s;
}
__global__ void k_colsum_r(const float* __restrict__ part, float* __restrict__ sums) {
  int col = blockIdx.x * 256 + threadIdx.x;
  float s = 0.f;
  for (int b = 0; b < 64; ++b) s += part[(size_t)b * 2048 + col];
  sums[col] = s;
}

// ---------------- pack xf [65536,200] -> B fragments (bf16) -----------------
// Bp[ks][ct][lane][j], ks in [0,2048), ct in [0,13): k = ks*32+(j>>3)*16+(lane>=16?8:0)+(j&7)
__global__ void k_pack1(const float* __restrict__ x, __bf16* __restrict__ Bp) {
  int ks = blockIdx.x, ct = blockIdx.y, lane = threadIdx.x;
  int c  = ct * 16 + (lane & 15);
  int k0 = ks * 32 + ((lane & 16) ? 8 : 0);
  v16bf v;
#pragma unroll
  for (int j = 0; j < 16; ++j) {
    float f = (c < 200) ? x[(size_t)(k0 + (j >> 3) * 16 + (j & 7)) * 200 + c] : 0.f;
    v[j] = (__bf16)f;
  }
  *(v16bf*)(Bp + (((size_t)ks * 13 + ct) * 32 + lane) * 16) = v;
}

// ---------------- GEMM1: C1[2048,208] += Q^T(2048x65536) * xf ---------------
// grid (64 m-blocks of 32 rows, 32 K-splits of 64 ksteps), one wave per block.
// Split-K accumulation via f32 atomicAdd (last-bit order jitter << bf16 noise).
__global__ void __launch_bounds__(32) k_gemm1(const float* __restrict__ Q,
                                              const __bf16* __restrict__ Bp,
                                              float* __restrict__ C1) {
  int lane  = threadIdx.x;
  int m0    = blockIdx.x * 32;
  int ksBeg = blockIdx.y * 64;
  int mlane = lane & 15;
  int khalf = (lane & 16) ? 8 : 0;     // A-fragment K-offset for upper lanes
  v8f acc[2][13];
#pragma unroll
  for (int t = 0; t < 2; ++t)
#pragma unroll
    for (int ct = 0; ct < 13; ++ct) acc[t][ct] = {};

  for (int ks = ksBeg; ks < ksBeg + 64; ++ks) {
    v16bf a[2];
#pragma unroll
    for (int t = 0; t < 2; ++t) {
      int m = m0 + t * 16 + mlane;
#pragma unroll
      for (int j = 0; j < 16; ++j) {
        int k = ks * 32 + (j >> 3) * 16 + khalf + (j & 7);
        a[t][j] = (__bf16)Q[(size_t)k * 2048 + m];   // transpose read of Q
      }
    }
    const __bf16* bb = Bp + (size_t)ks * 13 * 512 + lane * 16;
#pragma unroll
    for (int ct = 0; ct < 13; ++ct) {
      v16bf b = *(const v16bf*)(bb + ct * 512);
      acc[0][ct] = wmma_bf16(a[0], b, acc[0][ct]);
      acc[1][ct] = wmma_bf16(a[1], b, acc[1][ct]);
    }
  }
  int mo = (lane & 16) ? 8 : 0;
#pragma unroll
  for (int t = 0; t < 2; ++t)
#pragma unroll
    for (int ct = 0; ct < 13; ++ct)
#pragma unroll
      for (int v = 0; v < 8; ++v) {
        int m = m0 + t * 16 + mo + v;
        int c = ct * 16 + mlane;
        atomicAdd(&C1[m * 208 + c], acc[t][ct][v]);
      }
}

// ---------------- CSR build: per-row compacted neighbor list (deterministic)-
__global__ void k_csr(const float* __restrict__ adj, int* __restrict__ nbr,
                      float* __restrict__ nval, int* __restrict__ cnt) {
  __shared__ int lcnt[256];
  int n = blockIdx.x, tx = threadIdx.x;
  int j0 = tx * 8;                                  // 256 threads * 8 = 2048
  int c = 0;
  for (int t = 0; t < 8; ++t)
    c += (adj[(size_t)n * 2048 + j0 + t] != 0.f) ? 1 : 0;
  lcnt[tx] = c;
  __syncthreads();
  // inclusive Hillis-Steele scan
  for (int s = 1; s < 256; s <<= 1) {
    int a = (tx >= s) ? lcnt[tx - s] : 0;
    int t = lcnt[tx];
    __syncthreads();
    lcnt[tx] = t + a;
    __syncthreads();
  }
  int off = lcnt[tx] - c;                           // exclusive prefix
  for (int t = 0; t < 8; ++t) {
    float a = adj[(size_t)n * 2048 + j0 + t];
    if (a != 0.f) {
      if (off < CSR_CAP) {
        nbr [(size_t)n * CSR_CAP + off] = j0 + t;
        nval[(size_t)n * CSR_CAP + off] = a;
      }
      ++off;
    }
  }
  if (tx == 255) {
    int tot = lcnt[255];
    cnt[n] = tot < CSR_CAP ? tot : CSR_CAP;
  }
}

// ---------------- xn = BN * C1/sums ; xg = BN * xn (padded 208) -------------
__global__ void k_xn(const float* __restrict__ C1, const float* __restrict__ sums,
                     float* __restrict__ xn, float* __restrict__ xg) {
  int n = blockIdx.x, c = threadIdx.x;
  if (c >= 208) return;
  float v = 0.f;
  if (c < 200) v = BNC * C1[n * 208 + c] / sums[n];
  xn[n * 208 + c] = v;
  xg[n * 208 + c] = BNC * v;
}

// ---------------- small dense MM: Out = scale * (A @ W) ---------------------
__global__ void k_mm(const float* __restrict__ A, int lda,
                     const float* __restrict__ W, int ldw,
                     int Cin, int Cout, float scale,
                     float* __restrict__ Out, int ldo) {
  int n = blockIdx.x;
  const float* ar = A + (size_t)n * lda;
  for (int j = threadIdx.x; j < Cout; j += 256) {
    float acc = 0.f;
    for (int c = 0; c < Cin; ++c) acc += ar[c] * W[(size_t)c * ldw + j];
    Out[(size_t)n * ldo + j] = scale * acc;
  }
}

// ---------------- CSR adj @ M + bias (optional lrelu 0.01) ------------------
__global__ void k_adjmm(const int* __restrict__ nbr, const float* __restrict__ nval,
                        const int* __restrict__ cnt, const float* __restrict__ M,
                        int ldm, const float* __restrict__ bias,
                        int Cout, int ldo, int doAct, float* __restrict__ Out) {
  __shared__ int   sj[CSR_CAP];
  __shared__ float sv[CSR_CAP];
  int n = blockIdx.x, tx = threadIdx.x;
  int nc = cnt[n];
  for (int p = tx; p < nc; p += 256) {
    sj[p] = nbr [(size_t)n * CSR_CAP + p];
    sv[p] = nval[(size_t)n * CSR_CAP + p];
  }
  __syncthreads();
  for (int j = tx; j < Cout; j += 256) {
    float acc = bias[j];
    for (int p = 0; p < nc; ++p) acc += sv[p] * M[(size_t)sj[p] * ldm + j];
    if (doAct) acc = lrelu_(acc, 0.01f);
    Out[(size_t)n * ldo + j] = acc;
  }
}

// ---------------- residual add: xg += lrelu(BN*v, .01) ----------------------
__global__ void k_resadd(const float* __restrict__ v, float* __restrict__ xg) {
  int n = blockIdx.x, c = threadIdx.x;
  if (c < 200) xg[n * 208 + c] += lrelu_(BNC * v[n * 208 + c], 0.01f);
}

// ---------------- f1/f2 projections for a GAT head --------------------------
__global__ void k_f12(const float* __restrict__ Wh, int ld, const float* __restrict__ a,
                      int nh, float* __restrict__ f1, float* __restrict__ f2) {
  int n = blockIdx.x * 256 + threadIdx.x;
  if (n >= 2048) return;
  const float* r = Wh + (size_t)n * ld;
  float s1 = 0.f, s2 = 0.f;
  for (int c = 0; c < nh; ++c) { s1 += r[c] * a[c]; s2 += r[c] * a[nh + c]; }
  f1[n] = s1; f2[n] = s2;
}

// ---------------- masked softmax attention row (CSR), ELU output ------------
__global__ void k_att(const int* __restrict__ nbr, const int* __restrict__ cnt,
                      const float* __restrict__ f1, const float* __restrict__ f2,
                      const float* __restrict__ Wh, int ldWh, int nhid,
                      float* __restrict__ Out, int ldo, int outOff) {
  __shared__ int   sj[CSR_CAP];
  __shared__ float w [CSR_CAP];
  __shared__ float red[256];
  int n = blockIdx.x, tx = threadIdx.x;
  int nc = cnt[n];
  for (int p = tx; p < nc; p += 256) sj[p] = nbr[(size_t)n * CSR_CAP + p];
  __syncthreads();
  float f1n = f1[n];
  // max over masked lrelu(0.2) scores (self-loop guarantees nc >= 1)
  float lm = -3.0e38f;
  for (int p = tx; p < nc; p += 256)
    lm = fmaxf(lm, lrelu_(f1n + f2[sj[p]], 0.2f));
  red[tx] = lm; __syncthreads();
  for (int s = 128; s > 0; s >>= 1) { if (tx < s) red[tx] = fmaxf(red[tx], red[tx + s]); __syncthreads(); }
  float m = red[0]; __syncthreads();
  // exp weights + sum (fixed order: strided partials then tree)
  float ls = 0.f;
  for (int p = tx; p < nc; p += 256) {
    float ww = __expf(lrelu_(f1n + f2[sj[p]], 0.2f) - m);
    w[p] = ww;
    ls += ww;
  }
  red[tx] = ls; __syncthreads();
  for (int s = 128; s > 0; s >>= 1) { if (tx < s) red[tx] += red[tx + s]; __syncthreads(); }
  float inv = 1.f / red[0]; __syncthreads();
  // weighted neighbor sum, ELU
  for (int c = tx; c < nhid; c += 256) {
    float acc = 0.f;
    for (int p = 0; p < nc; ++p) acc += w[p] * Wh[(size_t)sj[p] * ldWh + c];
    float r = acc * inv;
    r = r > 0.f ? r : (__expf(r) - 1.f);
    Out[(size_t)n * ldo + outOff + c] = r;
  }
}

// ---------------- adaptive fusion + final bn/lrelu(0.2) ---------------------
__global__ void k_fuse(const float* __restrict__ vG, const float* __restrict__ vA,
                       const float* __restrict__ Wf, const float* __restrict__ bfp,
                       const float* __restrict__ cw, const float* __restrict__ cb,
                       float* __restrict__ yfin) {
  int n = blockIdx.x, c = threadIdx.x;  // 64 threads
  float g = lrelu_(BNC * vG[n * 64 + c], 0.01f);   // GCN_x
  float a = lrelu_(BNC * vA[n * 64 + c], 0.01f);   // GAT_x
  float mn = fminf(g, a), mx = fmaxf(g, a);
  float avg = g * Wf[c] + bfp[c] + a * Wf[64 + c] + bfp[64 + c];
  float s = g + a;
  float y = cw[0] * (mn + s) + cw[1] * (mx + s) + cw[2] * (avg + s) + cb[0];
  yfin[n * 64 + c] = lrelu_(BNC * y, 0.2f);
}

// ---------------- pack y [2048,64] -> B fragments (bf16) --------------------
__global__ void k_pack2(const float* __restrict__ y, __bf16* __restrict__ Bp) {
  int ks = blockIdx.x, ct = blockIdx.y, lane = threadIdx.x;
  int c  = ct * 16 + (lane & 15);
  int k0 = ks * 32 + ((lane & 16) ? 8 : 0);
  v16bf v;
#pragma unroll
  for (int j = 0; j < 16; ++j) v[j] = (__bf16)y[(size_t)(k0 + (j >> 3) * 16 + (j & 7)) * 64 + c];
  *(v16bf*)(Bp + (((size_t)ks * 4 + ct) * 32 + lane) * 16) = v;
}

// ---------------- GEMM2: out[65536,64] = Q(65536x2048) @ y ------------------
__global__ void __launch_bounds__(32) k_gemm2(const float* __restrict__ Q,
                                              const __bf16* __restrict__ Bp,
                                              float* __restrict__ Out) {
  int lane  = threadIdx.x;
  int i0    = blockIdx.x * 16;
  int mlane = lane & 15;
  int khalf = (lane & 16) ? 8 : 0;
  const float* qrow = Q + (size_t)(i0 + mlane) * 2048 + khalf;
  v8f acc[4];
#pragma unroll
  for (int t = 0; t < 4; ++t) acc[t] = {};
  for (int ks = 0; ks < 64; ++ks) {
    const float* p = qrow + ks * 32;
    float4 f0 = *(const float4*)(p);
    float4 f1 = *(const float4*)(p + 4);
    float4 f2 = *(const float4*)(p + 16);
    float4 f3 = *(const float4*)(p + 20);
    v16bf a;
    a[0]=(__bf16)f0.x; a[1]=(__bf16)f0.y; a[2]=(__bf16)f0.z; a[3]=(__bf16)f0.w;
    a[4]=(__bf16)f1.x; a[5]=(__bf16)f1.y; a[6]=(__bf16)f1.z; a[7]=(__bf16)f1.w;
    a[8]=(__bf16)f2.x; a[9]=(__bf16)f2.y; a[10]=(__bf16)f2.z; a[11]=(__bf16)f2.w;
    a[12]=(__bf16)f3.x; a[13]=(__bf16)f3.y; a[14]=(__bf16)f3.z; a[15]=(__bf16)f3.w;
    const __bf16* bb = Bp + (size_t)ks * 4 * 512 + lane * 16;
#pragma unroll
    for (int t = 0; t < 4; ++t) {
      v16bf b = *(const v16bf*)(bb + t * 512);
      acc[t] = wmma_bf16(a, b, acc[t]);
    }
  }
  int mo = (lane & 16) ? 8 : 0;
#pragma unroll
  for (int t = 0; t < 4; ++t)
#pragma unroll
    for (int v = 0; v < 8; ++v)
      Out[(size_t)(i0 + mo + v) * 64 + t * 16 + mlane] = acc[t][v];
}

// ===========================================================================
extern "C" void kernel_launch(void* const* d_in, const int* in_sizes, int n_in,
                              void* d_out, int out_size, void* d_ws, size_t ws_size,
                              hipStream_t stream) {
  const float* x      = (const float*)d_in[0];
  const float* adj    = (const float*)d_in[1];
  const float* Q      = (const float*)d_in[2];
  const float* g1a_W  = (const float*)d_in[3];
  const float* g1a_b  = (const float*)d_in[4];
  const float* g1b_W  = (const float*)d_in[5];
  const float* g1b_b  = (const float*)d_in[6];
  const float* g2a_W  = (const float*)d_in[7];
  const float* g2a_b  = (const float*)d_in[8];
  const float* g2b_W  = (const float*)d_in[9];
  const float* g2b_b  = (const float*)d_in[10];
  const float* gat_W  = (const float*)d_in[11];
  const float* gat_a  = (const float*)d_in[12];
  const float* out_W  = (const float*)d_in[13];
  const float* out_a  = (const float*)d_in[14];
  const float* Wf     = (const float*)d_in[15];
  const float* bfp    = (const float*)d_in[16];
  const float* conv_w = (const float*)d_in[17];
  const float* conv_b = (const float*)d_in[18];

  float* W0    = (float*)d_ws;
  float* sums  = W0;                    // 2048
  float* C1    = W0 + 2048;             // 2048*208
  float* xn    = C1 + 425984;
  float* xg    = xn + 425984;
  float* b200a = xg + 425984;
  float* b200b = b200a + 425984;
  float* t128a = b200b + 425984;        // 2048*128
  float* t128b = t128a + 262144;
  float* c64a  = t128b + 262144;        // 2048*64
  float* c64b  = c64a + 131072;
  float* f1b   = c64b + 131072;         // 5*2048
  float* f2b   = f1b + 10240;
  float* part  = f2b + 10240;           // 64*2048 colsum partials
  float* uni   = part + 131072;         // union region (Bp1 dead after gemm1)
  __bf16* Bp1  = (__bf16*)uni;          // 2048*13*512 bf16 = 27.3MB extent
  float* Wh    = uni;                   // 4*2048*128 (aliases Bp1, written later)
  float* hcat  = Wh + 4 * 262144;       // 2048*512
  float* whO   = hcat + 1048576;        // 2048*64
  float* attO  = whO + 131072;
  float* yfin  = attO + 131072;
  __bf16* Bp2  = (__bf16*)(yfin + 131072);      // 64*4*512 bf16 (65536 floats)
  int*   nbr   = (int*)(yfin + 131072 + 65536); // 2048*640 ints
  float* nval  = (float*)(nbr + 2048 * CSR_CAP);
  int*   ncnt  = (int*)(nval + 2048 * CSR_CAP); // 2048 ints (fits in Bp1 extent)

  // zero sums + C1 (428032 floats, contiguous)
  k_init<<<1672, 256, 0, stream>>>(W0);
  k_colsum_p<<<dim3(8, 64), 256, 0, stream>>>(Q, part);
  k_colsum_r<<<8, 256, 0, stream>>>(part, sums);
  k_pack1<<<dim3(2048, 13), 32, 0, stream>>>(x, Bp1);
  k_gemm1<<<dim3(64, 32), 32, 0, stream>>>(Q, Bp1, C1);
  k_csr<<<2048, 256, 0, stream>>>(adj, nbr, nval, ncnt);   // after gemm1 (aliases Bp1)
  k_xn<<<2048, 256, 0, stream>>>(C1, sums, xn, xg);

  // ---- GCN block 1 ----
  k_mm<<<2048, 256, 0, stream>>>(xg, 208, g1a_W, 128, 200, 128, BNC, t128a, 128);
  k_adjmm<<<2048, 256, 0, stream>>>(nbr, nval, ncnt, t128a, 128, g1a_b, 128, 128, 1, t128b);
  k_mm<<<2048, 256, 0, stream>>>(t128b, 128, g1b_W, 200, 128, 200, BNC, b200a, 208);
  k_adjmm<<<2048, 256, 0, stream>>>(nbr, nval, ncnt, b200a, 208, g1b_b, 200, 208, 0, b200b);
  k_resadd<<<2048, 256, 0, stream>>>(b200b, xg);
  // ---- GCN block 2 ----
  k_mm<<<2048, 256, 0, stream>>>(xg, 208, g2a_W, 128, 200, 128, BNC, t128a, 128);
  k_adjmm<<<2048, 256, 0, stream>>>(nbr, nval, ncnt, t128a, 128, g2a_b, 128, 128, 1, t128b);
  k_mm<<<2048, 256, 0, stream>>>(t128b, 128, g2b_W, 64, 128, 64, BNC, c64a, 64);
  k_adjmm<<<2048, 256, 0, stream>>>(nbr, nval, ncnt, c64a, 64, g2b_b, 64, 64, 0, c64b);

  // ---- GAT heads ----
  for (int h = 0; h < 4; ++h) {
    k_mm<<<2048, 256, 0, stream>>>(xn, 208, gat_W + h * 25600, 128, 200, 128, 1.0f,
                                   Wh + h * 262144, 128);
    k_f12<<<8, 256, 0, stream>>>(Wh + h * 262144, 128, gat_a + h * 256, 128,
                                 f1b + h * 2048, f2b + h * 2048);
    k_att<<<2048, 256, 0, stream>>>(nbr, ncnt, f1b + h * 2048, f2b + h * 2048,
                                    Wh + h * 262144, 128, 128, hcat, 512, h * 128);
  }
  // ---- GAT output attention ----
  k_mm<<<2048, 256, 0, stream>>>(hcat, 512, out_W, 64, 512, 64, 1.0f, whO, 64);
  k_f12<<<8, 256, 0, stream>>>(whO, 64, out_a, 64, f1b + 8192, f2b + 8192);
  k_att<<<2048, 256, 0, stream>>>(nbr, ncnt, f1b + 8192, f2b + 8192, whO, 64, 64, attO, 64, 0);

  // ---- fusion + final projection ----
  k_fuse<<<2048, 64, 0, stream>>>(c64b, attO, Wf, bfp, conv_w, conv_b, yfin);
  k_pack2<<<dim3(64, 4), 32, 0, stream>>>(yfin, Bp2);
  k_gemm2<<<4096, 32, 0, stream>>>(Q, Bp2, (float*)d_out);
}
